// GraphConvolution_14190571946144
// MI455X (gfx1250) — matrix operand order repaired and checked by reading, not compiled, not run on hardware
//
#include <hip/hip_runtime.h>
#include <hip/hip_bf16.h>
#include <math.h>

typedef __attribute__((ext_vector_type(16))) _Float16 v16h;
typedef __attribute__((ext_vector_type(8)))  float    v8f;

#define D 128
#define COLSUM_PARTS 64
#define WPACK_HALVES (3 * 4 * 8 * 32 * 16)   // 49152 halves = 96 KB
#define WPACK_CHUNKS (WPACK_HALVES * 2 / 16) // 6144 x 16B

// ---------------------------------------------------------------------------
// Repack W[128][128] f32 (row=K/in, col=N/out) into per-wave WMMA B layout:
// wpack[(((w*4+kt)*8+nt)*32+lane)*16 + i] =
//     (half) W[kt*32 + (i<8 ? 8*(lane>=16) + i : 16 + 8*(lane>=16) + i-8)][nt*16 + lane%16]
// ---------------------------------------------------------------------------
__global__ void prep_weights_kernel(const float* __restrict__ w1,
                                    const float* __restrict__ w2,
                                    const float* __restrict__ w3,
                                    _Float16* __restrict__ wpack) {
    int t = blockIdx.x * blockDim.x + threadIdx.x;      // 0 .. 3071
    if (t >= 3 * 4 * 8 * 32) return;
    int lane = t & 31;
    int nt   = (t >> 5) & 7;
    int kt   = (t >> 8) & 3;
    int w    = t >> 10;
    const float* W = (w == 0) ? w1 : (w == 1) ? w2 : w3;
    int col = nt * 16 + (lane & 15);
    int hi  = lane >> 4;
    _Float16* dst = wpack + (size_t)t * 16;
#pragma unroll
    for (int i = 0; i < 16; ++i) {
        int k = kt * 32 + (i < 8 ? hi * 8 + i : 16 + hi * 8 + (i - 8));
        dst[i] = (_Float16)W[k * D + col];
    }
}

// ---------------------------------------------------------------------------
// H_w = tanh(X @ W_w) for w = 0..2, one 16-row tile per wave, WMMA f16->f32.
// All three packed weight matrices (96 KB) are staged into LDS once per block
// via gfx1250 async global->LDS copies (ASYNCcnt), then read via ds_load.
// ---------------------------------------------------------------------------
__global__ __launch_bounds__(256) void gemm3_tanh_kernel(
    const float* __restrict__ X, const _Float16* __restrict__ wpack,
    float* __restrict__ H0, float* __restrict__ H1, float* __restrict__ H2,
    int N) {
    __shared__ _Float16 sW[WPACK_HALVES];               // 96 KB

    // ---- async-copy wpack (global) -> sW (LDS), 16B per lane per issue ----
    {
        unsigned    ldsbase = (unsigned)(size_t)&sW[0]; // low 32b == LDS offset
        const char* gbase   = (const char*)wpack;
        for (int i = threadIdx.x; i < WPACK_CHUNKS; i += 256) {
            unsigned    l = ldsbase + (unsigned)i * 16;
            const char* g = gbase + (size_t)i * 16;
            asm volatile("global_load_async_to_lds_b128 %0, %1, off"
                         :: "v"(l), "v"(g) : "memory");
        }
        asm volatile("s_wait_asynccnt 0x0" ::: "memory");
    }
    __syncthreads();

    int waveInBlock = threadIdx.x >> 5;
    int lane        = threadIdx.x & 31;
    int tile        = blockIdx.x * 8 + waveInBlock;     // 16-row tile
    int rowbase     = tile * 16;
    if (rowbase < N) {                                  // uniform per wave
        int lane16 = lane & 15;
        int hi     = lane >> 4;

        // ---- load A tiles (16x32 f16), 4 K-tiles covering K=0..127 --------
        v16h A[4];
        const float* xrow = X + (size_t)(rowbase + lane16) * D;
#pragma unroll
        for (int kt = 0; kt < 4; ++kt) {
            int k0 = kt * 32 + (hi ? 8 : 0);            // lanes>=16 hold K+8
            const float4* p0 = (const float4*)(xrow + k0);
            const float4* p1 = (const float4*)(xrow + k0 + 16);
            float4 f0 = p0[0], f1 = p0[1], f2 = p1[0], f3 = p1[1];
            union { v16h v; _Float16 e[16]; } u;
            u.e[0]  = (_Float16)f0.x; u.e[1]  = (_Float16)f0.y;
            u.e[2]  = (_Float16)f0.z; u.e[3]  = (_Float16)f0.w;
            u.e[4]  = (_Float16)f1.x; u.e[5]  = (_Float16)f1.y;
            u.e[6]  = (_Float16)f1.z; u.e[7]  = (_Float16)f1.w;
            u.e[8]  = (_Float16)f2.x; u.e[9]  = (_Float16)f2.y;
            u.e[10] = (_Float16)f2.z; u.e[11] = (_Float16)f2.w;
            u.e[12] = (_Float16)f3.x; u.e[13] = (_Float16)f3.y;
            u.e[14] = (_Float16)f3.z; u.e[15] = (_Float16)f3.w;
            A[kt] = u.v;
        }

#pragma unroll
        for (int w = 0; w < 3; ++w) {
            v8f acc[8] = {};
#pragma unroll
            for (int nt = 0; nt < 8; ++nt) {
#pragma unroll
                for (int kt = 0; kt < 4; ++kt) {
                    const v16h b = *(const v16h*)(sW +
                        (size_t)(((w * 4 + kt) * 8 + nt) * 32 + lane) * 16);
                    acc[nt] = __builtin_amdgcn_wmma_f32_16x16x32_f16(
                        false, A[kt], false, b, (short)0, acc[nt],
                        false, false);
                }
            }
            float* Hd = (w == 0) ? H0 : (w == 1) ? H1 : H2;
#pragma unroll
            for (int nt = 0; nt < 8; ++nt) {
#pragma unroll
                for (int r = 0; r < 8; ++r) {
                    int m = r + (hi ? 8 : 0);
                    Hd[(size_t)(rowbase + m) * D + nt * 16 + lane16] =
                        tanhf(acc[nt][r]);
                }
            }
        }
    }
}

// ---------------------------------------------------------------------------
// out[row] += val * in[col]  (one edge per wave: 32 lanes x float4 = 128 cols)
// ---------------------------------------------------------------------------
__global__ void spmm_kernel(const int* __restrict__ erow,
                            const int* __restrict__ ecol,
                            const float* __restrict__ eval,
                            const float* __restrict__ in,
                            float* __restrict__ out, int E) {
    long long t = (long long)blockIdx.x * blockDim.x + threadIdx.x;
    int e    = (int)(t >> 5);
    int lane = (int)(t & 31);
    if (e >= E) return;
    int   r = erow[e];
    int   c = ecol[e];
    float v = eval[e];
    float4 x = ((const float4*)(in + (size_t)c * D))[lane];
    float* dst = out + (size_t)r * D + lane * 4;
    atomicAdd(dst + 0, v * x.x);
    atomicAdd(dst + 1, v * x.y);
    atomicAdd(dst + 2, v * x.z);
    atomicAdd(dst + 3, v * x.w);
}

__global__ void zero_kernel(float4* __restrict__ p, int n4) {
    int i = blockIdx.x * blockDim.x + threadIdx.x;
    if (i < n4) p[i] = make_float4(0.f, 0.f, 0.f, 0.f);
}

// ---------------------------------------------------------------------------
// Partial column sums of the three feats buffers: partial[(o*PARTS+part)*128+d]
// ---------------------------------------------------------------------------
__global__ void colsum_kernel(const float* __restrict__ f1,
                              const float* __restrict__ f2,
                              const float* __restrict__ f3,
                              float* __restrict__ partial, int N) {
    const float* f = (blockIdx.y == 0) ? f1 : (blockIdx.y == 1) ? f2 : f3;
    int part = blockIdx.x;
    int d    = threadIdx.x;
    int chunk = (N + COLSUM_PARTS - 1) / COLSUM_PARTS;
    int n0 = part * chunk;
    int n1 = n0 + chunk; if (n1 > N) n1 = N;
    float s = 0.f;
    for (int n = n0; n < n1; ++n) s += f[(size_t)n * D + d];
    partial[((size_t)blockIdx.y * COLSUM_PARTS + part) * D + d] = s;
}

// ---------------------------------------------------------------------------
// m[o] = mean + b_o ; a = softmax_o(relu(m@fc1+fc1_b)@fc2+fc2_b);
// c[d] = sum_o a[o,d]*b_o[d].  Single block of 128 threads.
// ---------------------------------------------------------------------------
__global__ void attn_kernel(const float* __restrict__ partial,
                            const float* __restrict__ b1,
                            const float* __restrict__ b2,
                            const float* __restrict__ b3,
                            const float* __restrict__ fc1_w,
                            const float* __restrict__ fc1_b,
                            const float* __restrict__ fc2_w,
                            const float* __restrict__ fc2_b,
                            float* __restrict__ a_out,
                            float* __restrict__ c_out, int N) {
    __shared__ float m[3][D];
    __shared__ float r[3][32];
    int d = threadIdx.x;
    const float* bs[3] = {b1, b2, b3};
    for (int o = 0; o < 3; ++o) {
        float s = 0.f;
        for (int p = 0; p < COLSUM_PARTS; ++p)
            s += partial[((size_t)o * COLSUM_PARTS + p) * D + d];
        m[o][d] = s / (float)N + bs[o][d];
    }
    __syncthreads();
    if (d < 96) {
        int o = d >> 5, j = d & 31;
        float s = fc1_b[j];
        for (int k = 0; k < D; ++k) s += m[o][k] * fc1_w[k * 32 + j];
        r[o][j] = fmaxf(s, 0.f);
    }
    __syncthreads();
    float l0 = fc2_b[d], l1 = l0, l2 = l0;
    for (int j = 0; j < 32; ++j) {
        float w = fc2_w[j * D + d];
        l0 += r[0][j] * w; l1 += r[1][j] * w; l2 += r[2][j] * w;
    }
    float mx = fmaxf(l0, fmaxf(l1, l2));
    float e0 = expf(l0 - mx), e1 = expf(l1 - mx), e2 = expf(l2 - mx);
    float inv = 1.f / (e0 + e1 + e2);
    float a0 = e0 * inv, a1 = e1 * inv, a2 = e2 * inv;
    a_out[d]         = a0;
    a_out[D + d]     = a1;
    a_out[2 * D + d] = a2;
    c_out[d] = a0 * b1[d] + a1 * b2[d] + a2 * b3[d];
}

// ---------------------------------------------------------------------------
// out[n,d] = a0[d]*f1 + a1[d]*f2 + a2[d]*f3 + c[d]   (float4 per thread)
// ---------------------------------------------------------------------------
__global__ void combine_kernel(const float* __restrict__ f1,
                               const float* __restrict__ f2,
                               const float* __restrict__ f3,
                               const float* __restrict__ a,
                               const float* __restrict__ c,
                               float* __restrict__ out, int N) {
    long long idx = (long long)blockIdx.x * blockDim.x + threadIdx.x;
    long long total = (long long)N * (D / 4);
    if (idx >= total) return;
    size_t base = (size_t)idx * 4;
    int d = (int)(base & (D - 1));
    float4 x1 = *(const float4*)(f1 + base);
    float4 x2 = *(const float4*)(f2 + base);
    float4 x3 = *(const float4*)(f3 + base);
    float4 A0 = *(const float4*)(a + d);
    float4 A1 = *(const float4*)(a + D + d);
    float4 A2 = *(const float4*)(a + 2 * D + d);
    float4 C  = *(const float4*)(c + d);
    float4 o;
    o.x = A0.x * x1.x + A1.x * x2.x + A2.x * x3.x + C.x;
    o.y = A0.y * x1.y + A1.y * x2.y + A2.y * x3.y + C.y;
    o.z = A0.z * x1.z + A1.z * x2.z + A2.z * x3.z + C.z;
    o.w = A0.w * x1.w + A1.w * x2.w + A2.w * x3.w + C.w;
    *(float4*)(out + base) = o;
}

// ---------------------------------------------------------------------------
extern "C" void kernel_launch(void* const* d_in, const int* in_sizes, int n_in,
                              void* d_out, int out_size, void* d_ws, size_t ws_size,
                              hipStream_t stream) {
    const float* features = (const float*)d_in[0];
    const int*   edge_row = (const int*)d_in[1];
    const int*   edge_col = (const int*)d_in[2];
    const float* edge_val = (const float*)d_in[3];
    const float* w1 = (const float*)d_in[4];
    const float* b1 = (const float*)d_in[5];
    const float* w2 = (const float*)d_in[6];
    const float* b2 = (const float*)d_in[7];
    const float* w3 = (const float*)d_in[8];
    const float* b3 = (const float*)d_in[9];
    const float* fc1_w = (const float*)d_in[10];
    const float* fc1_b = (const float*)d_in[11];
    const float* fc2_w = (const float*)d_in[12];
    const float* fc2_b = (const float*)d_in[13];

    const int N = in_sizes[0] / D;
    const int E = in_sizes[1];

    // ---- workspace layout (floats) ----------------------------------------
    const size_t NB = (size_t)N * D;                    // one node-feature buf
    float* ws = (float*)d_ws;
    float* B0 = ws;                                     // gemm H1 -> feats2
    float* B1 = ws + 1 * NB;                            // gemm H2 -> tmp
    float* B2 = ws + 2 * NB;                            // gemm H3 -> feats3
    float* B3 = ws + 3 * NB;                            // feats1
    float* B4 = ws + 4 * NB;                            // tmp
    float* partial = ws + 5 * NB;                       // 3*64*128
    float* a_buf   = partial + 3 * COLSUM_PARTS * D;    // 3*128
    float* c_buf   = a_buf + 3 * D;                     // 128
    _Float16* wpack = (_Float16*)(c_buf + D);           // 3*4*8*32*16 halves

    const int n4 = (int)(NB / 4);
    const int zeroBlocks = (n4 + 255) / 256;
    const int spmmBlocks = (int)(((long long)E * 32 + 255) / 256);

    // 1) repack weights to WMMA-B f16 layout
    prep_weights_kernel<<<12, 256, 0, stream>>>(w1, w2, w3, wpack);

    // 2) H_k = tanh(X @ W_k), k=1..3  (WMMA, weights staged in LDS async)
    int tiles = N / 16;
    gemm3_tanh_kernel<<<(tiles + 7) / 8, 256, 0, stream>>>(
        features, wpack, B0, B1, B2, N);

    // 3) branch 1: feats1 = S*H1   (B0 -> B3)
    zero_kernel<<<zeroBlocks, 256, 0, stream>>>((float4*)B3, n4);
    spmm_kernel<<<spmmBlocks, 256, 0, stream>>>(edge_row, edge_col, edge_val, B0, B3, E);

    // 4) branch 2: feats2 = S^2*H2 (B1 -> B4 -> B0)
    zero_kernel<<<zeroBlocks, 256, 0, stream>>>((float4*)B4, n4);
    spmm_kernel<<<spmmBlocks, 256, 0, stream>>>(edge_row, edge_col, edge_val, B1, B4, E);
    zero_kernel<<<zeroBlocks, 256, 0, stream>>>((float4*)B0, n4);
    spmm_kernel<<<spmmBlocks, 256, 0, stream>>>(edge_row, edge_col, edge_val, B4, B0, E);

    // 5) branch 3: feats3 = S^3*H3 (B2 -> B4 -> B1 -> B2)
    zero_kernel<<<zeroBlocks, 256, 0, stream>>>((float4*)B4, n4);
    spmm_kernel<<<spmmBlocks, 256, 0, stream>>>(edge_row, edge_col, edge_val, B2, B4, E);
    zero_kernel<<<zeroBlocks, 256, 0, stream>>>((float4*)B1, n4);
    spmm_kernel<<<spmmBlocks, 256, 0, stream>>>(edge_row, edge_col, edge_val, B4, B1, E);
    zero_kernel<<<zeroBlocks, 256, 0, stream>>>((float4*)B2, n4);
    spmm_kernel<<<spmmBlocks, 256, 0, stream>>>(edge_row, edge_col, edge_val, B1, B2, E);

    // feats: f1=B3, f2=B0, f3=B2
    // 6) column means -> tiny MLP -> softmax weights a, bias-combine c
    colsum_kernel<<<dim3(COLSUM_PARTS, 3), D, 0, stream>>>(B3, B0, B2, partial, N);
    attn_kernel<<<1, D, 0, stream>>>(partial, b1, b2, b3,
                                     fc1_w, fc1_b, fc2_w, fc2_b,
                                     a_buf, c_buf, N);

    // 7) out = sum_o a[o] * feats_o + c
    long long total4 = (long long)N * (D / 4);
    combine_kernel<<<(int)((total4 + 255) / 256), 256, 0, stream>>>(
        B3, B0, B2, a_buf, c_buf, (float*)d_out, N);
}
